// PtrNet_90933047591077
// MI455X (gfx1250) — compile-verified, matrix operand order checked by reading, not run on hardware
//
#include <hip/hip_runtime.h>

// Problem constants (from reference): B=512, G=128, E=256, H=512
constexpr int kB  = 512;
constexpr int kG  = 128;
constexpr int kE  = 256;
constexpr int kH  = 512;
constexpr int kH4 = 4 * kH;      // 2048
constexpr int kXH = kE + kH;     // 768  (logical K of the gates GEMM)

typedef __attribute__((ext_vector_type(16))) __bf16        v16bf;
typedef __attribute__((ext_vector_type(8)))  float         v8f;
typedef __attribute__((ext_vector_type(4)))  unsigned int  u32x4;

// ---------------------------------------------------------------- helpers
__device__ __forceinline__ unsigned short f2bf(float f) {
  unsigned int x = __float_as_uint(f);
  x += 0x7FFFu + ((x >> 16) & 1u);              // round-to-nearest-even
  return (unsigned short)(x >> 16);
}
__device__ __forceinline__ float bf2f(unsigned short u) {
  return __uint_as_float(((unsigned int)u) << 16);
}
__device__ __forceinline__ float sigmoidf(float x) {
  return 1.0f / (1.0f + __expf(-x));
}

// Per-lane fragment: 8 bf16 at p (K+0..7) and 8 bf16 at p+16 (K+16..23);
// hi-half lanes pass p already offset by +8, matching the CDNA5 16-bit
// 16x32 A/B VGPR layout.
__device__ __forceinline__ v16bf load_frag(const unsigned short* p) {
  union { u32x4 q[2]; v16bf v; } r;
  r.q[0] = *reinterpret_cast<const u32x4*>(p);
  r.q[1] = *reinterpret_cast<const u32x4*>(p + 16);
  return r.v;
}

// ---------------------------------------------------------------- WMMA GEMM
// C[M,N] = [A1 | A2][M, K1+K2] (bf16) * W[N, K1+K2]^T (bf16 row-major).
// Split-K-source form: A1 has lda1 (K1 cols), A2 has lda2 (K2 cols, may be
// null when K2==0). Wave tile 16x64 (4 accumulators, A fragment reused 4x).
// 256 threads = 8 waves -> block tile 64x128. grid = (M/64, N/128).
template <bool BF16_OUT>
__global__ void gemm_bt_wmma(const unsigned short* __restrict__ A1, int lda1, int K1,
                             const unsigned short* __restrict__ A2, int lda2, int K2,
                             const unsigned short* __restrict__ W, int ldw,
                             void* __restrict__ Cout, int ldc)
{
  const int wave  = threadIdx.x >> 5;
  const int lane  = threadIdx.x & 31;
  const int wm    = wave & 3;            // 4 M tiles
  const int wn    = wave >> 2;           // 2 N groups of 64
  const int tileM = blockIdx.x * 64 + wm * 16;
  const int tileN = blockIdx.y * 128 + wn * 64;
  const int lid   = lane & 15;
  const int hi    = lane >> 4;

  const unsigned short* pw[4];
#pragma unroll
  for (int nt = 0; nt < 4; ++nt)
    pw[nt] = W + (size_t)(tileN + nt * 16 + lid) * ldw + hi * 8;

  v8f acc[4] = {{}, {}, {}, {}};

  // ---- K segment 1 (x part)
  {
    const unsigned short* pa = A1 + (size_t)(tileM + lid) * lda1 + hi * 8;
    for (int k = 0; k < K1; k += 32) {
      if (k + 32 < K1) {
        __builtin_prefetch(pa + k + 32, 0, 1);
        __builtin_prefetch(pw[0] + k + 32, 0, 1);
      }
      v16bf a = load_frag(pa + k);
#pragma unroll
      for (int nt = 0; nt < 4; ++nt) {
        v16bf b = load_frag(pw[nt] + k);
        acc[nt] = __builtin_amdgcn_wmma_f32_16x16x32_bf16(
            false, a, false, b, (short)0, acc[nt], false, false);
      }
    }
  }

  // ---- K segment 2 (h part); weight fragment offset by K1
  if (K2 > 0) {
    const unsigned short* pa = A2 + (size_t)(tileM + lid) * lda2 + hi * 8;
    for (int k = 0; k < K2; k += 32) {
      if (k + 32 < K2) {
        __builtin_prefetch(pa + k + 32, 0, 1);
        __builtin_prefetch(pw[0] + K1 + k + 32, 0, 1);
      }
      v16bf a = load_frag(pa + k);
#pragma unroll
      for (int nt = 0; nt < 4; ++nt) {
        v16bf b = load_frag(pw[nt] + K1 + k);
        acc[nt] = __builtin_amdgcn_wmma_f32_16x16x32_bf16(
            false, a, false, b, (short)0, acc[nt], false, false);
      }
    }
  }

  // C/D layout: VGPR r, lanes 0-15 -> (M=r, N=lid); lanes 16-31 -> (M=r+8, N=lid)
#pragma unroll
  for (int nt = 0; nt < 4; ++nt) {
    const int col = tileN + nt * 16 + lid;
    size_t base = (size_t)(tileM + hi * 8) * (size_t)ldc + col;
#pragma unroll
    for (int r = 0; r < 8; ++r) {
      if (BF16_OUT)
        reinterpret_cast<unsigned short*>(Cout)[base + (size_t)r * ldc] = f2bf(acc[nt][r]);
      else
        reinterpret_cast<float*>(Cout)[base + (size_t)r * ldc] = acc[nt][r];
    }
  }
}

// ---------------------------------------------------------------- pointwise
// i,f,g,o split of gates; updates c; writes h as bf16 to h_b (and enc_out).
__global__ void lstm_pointwise(const float* __restrict__ gates,
                               const float* __restrict__ bih,
                               const float* __restrict__ bhh,
                               float* __restrict__ c,
                               unsigned short* __restrict__ h_b,
                               unsigned short* __restrict__ enc_slice,
                               size_t enc_stride)
{
  int idx = blockIdx.x * blockDim.x + threadIdx.x;
  if (idx >= kB * kH) return;
  int b = idx / kH, j = idx - b * kH;
  const float* g = gates + (size_t)b * kH4;
  float gi = g[j]          + bih[j]          + bhh[j];
  float gf = g[kH + j]     + bih[kH + j]     + bhh[kH + j];
  float gg = g[2*kH + j]   + bih[2*kH + j]   + bhh[2*kH + j];
  float go = g[3*kH + j]   + bih[3*kH + j]   + bhh[3*kH + j];
  float cc = sigmoidf(gf) * c[idx] + sigmoidf(gi) * tanhf(gg);
  float hh = sigmoidf(go) * tanhf(cc);
  c[idx] = cc;
  unsigned short hb = f2bf(hh);
  h_b[idx] = hb;
  if (enc_slice) enc_slice[(size_t)b * enc_stride + j] = hb;
}

// ---------------------------------------------------------------- packing
__global__ void pack_wcat(const float* __restrict__ Wih,
                          const float* __restrict__ Whh,
                          unsigned short* __restrict__ out)   // (4H, E+H)
{
  int idx = blockIdx.x * blockDim.x + threadIdx.x;
  if (idx >= kH4 * kXH) return;
  int n = idx / kXH, k = idx - n * kXH;
  float v = (k < kE) ? Wih[(size_t)n * kE + k]
                     : Whh[(size_t)n * kH + (k - kE)];
  out[idx] = f2bf(v);
}

__global__ void f32_to_bf16_k(const float* __restrict__ src,
                              unsigned short* __restrict__ dst, int n)
{
  int idx = blockIdx.x * blockDim.x + threadIdx.x;
  if (idx < n) dst[idx] = f2bf(src[idx]);
}

__global__ void fill_f32(float* p, float v, int n) {
  int idx = blockIdx.x * blockDim.x + threadIdx.x;
  if (idx < n) p[idx] = v;
}
__global__ void fill_u16(unsigned short* p, unsigned short v, int n) {
  int idx = blockIdx.x * blockDim.x + threadIdx.x;
  if (idx < n) p[idx] = v;
}

// emb[b,t,e] = pad ? 0 : node_emb[state[b,t], e]  (bf16)
__global__ void embed_kernel(const float* __restrict__ state,
                             const float* __restrict__ node_emb,
                             unsigned short* __restrict__ emb)
{
  int idx = blockIdx.x * blockDim.x + threadIdx.x;
  if (idx >= kB * kG * kE) return;
  int b = idx / (kG * kE);
  int rem = idx - b * kG * kE;
  int t = rem / kE, e = rem - t * kE;
  float sv = state[b * kG + t];
  float val = (sv < 0.0f) ? 0.0f : node_emb[(size_t)((int)sv) * kE + e];
  emb[idx] = f2bf(val);
}

// ---------------------------------------------------------------- attention
// One block per batch row b; 128 threads (one per node gi).
// energies -> softmax -> mask -> renorm -> argmax (first-max) -> gather x.
__global__ void attention_step(const unsigned short* __restrict__ ref_proj, // (B,G,H) bf16
                               const float* __restrict__ q,                 // (B,H)
                               const float* __restrict__ vvec,              // (H)
                               float* __restrict__ mask,                    // (B,G)
                               const float* __restrict__ action_emb,        // (G,E)
                               unsigned short* __restrict__ x_b,            // (B,E) bf16
                               float* __restrict__ out_actions,             // (B*G)
                               float* __restrict__ out_attn,                // (B,G,G)
                               int t)
{
  __shared__ float sq[kH];
  __shared__ float sv[kH];
  __shared__ float red[kG];
  __shared__ int   ridx[kG];

  const int b  = blockIdx.x;
  const int gi = threadIdx.x;   // 0..127

  for (int j = gi; j < kH; j += kG) {
    sq[j] = q[(size_t)b * kH + j];
    sv[j] = vvec[j];
  }
  __syncthreads();

  const unsigned short* rp = ref_proj + ((size_t)b * kG + gi) * kH;
  float e = 0.0f;
  for (int j = 0; j < kH; j += 8) {
    u32x4 pk = *reinterpret_cast<const u32x4*>(rp + j);
    const unsigned short* pu = reinterpret_cast<const unsigned short*>(&pk);
#pragma unroll
    for (int u = 0; u < 8; ++u)
      e += sv[j + u] * tanhf(bf2f(pu[u]) + sq[j + u]);
  }

  // softmax (max-stabilized)
  red[gi] = e; __syncthreads();
#pragma unroll
  for (int off = kG / 2; off > 0; off >>= 1) {
    if (gi < off) red[gi] = fmaxf(red[gi], red[gi + off]);
    __syncthreads();
  }
  const float mx = red[0]; __syncthreads();
  float p = __expf(e - mx);
  red[gi] = p; __syncthreads();
#pragma unroll
  for (int off = kG / 2; off > 0; off >>= 1) {
    if (gi < off) red[gi] += red[gi + off];
    __syncthreads();
  }
  const float denom = red[0]; __syncthreads();

  float mp = (p / denom) * mask[(size_t)b * kG + gi];
  red[gi] = mp; __syncthreads();
#pragma unroll
  for (int off = kG / 2; off > 0; off >>= 1) {
    if (gi < off) red[gi] += red[gi + off];
    __syncthreads();
  }
  const float msum = red[0]; __syncthreads();
  mp = mp / msum;

  out_attn[((size_t)b * kG + t) * kG + gi] = mp;

  // argmax with first-index tie-break
  red[gi] = mp; ridx[gi] = gi; __syncthreads();
#pragma unroll
  for (int off = kG / 2; off > 0; off >>= 1) {
    if (gi < off) {
      float a = red[gi], c2 = red[gi + off];
      int ia = ridx[gi], ic = ridx[gi + off];
      if (c2 > a || (c2 == a && ic < ia)) { red[gi] = c2; ridx[gi] = ic; }
    }
    __syncthreads();
  }
  const int topi = ridx[0];

  if (gi == 0) {
    mask[(size_t)b * kG + topi]     = 0.0f;
    out_actions[(size_t)b * kG + t] = (float)topi;
  }
  for (int e0 = gi; e0 < kE; e0 += kG)
    x_b[(size_t)b * kE + e0] = f2bf(action_emb[(size_t)topi * kE + e0]);
}

// ---------------------------------------------------------------- driver
extern "C" void kernel_launch(void* const* d_in, const int* in_sizes, int n_in,
                              void* d_out, int out_size, void* d_ws, size_t ws_size,
                              hipStream_t stream)
{
  (void)in_sizes; (void)n_in; (void)out_size; (void)ws_size;

  const float* state      = (const float*)d_in[0];
  const float* node_emb   = (const float*)d_in[1];
  const float* action_emb = (const float*)d_in[2];
  const float* Wih_e      = (const float*)d_in[3];
  const float* Whh_e      = (const float*)d_in[4];
  const float* bih_e      = (const float*)d_in[5];
  const float* bhh_e      = (const float*)d_in[6];
  const float* Wih_d      = (const float*)d_in[7];
  const float* Whh_d      = (const float*)d_in[8];
  const float* bih_d      = (const float*)d_in[9];
  const float* bhh_d      = (const float*)d_in[10];
  const float* W_ref      = (const float*)d_in[11];
  const float* W_q        = (const float*)d_in[12];
  const float* vvec       = (const float*)d_in[13];

  // workspace carve-out (~182 MB total)
  char* ws = (char*)d_ws;
  size_t off = 0;
  auto alloc = [&](size_t bytes) -> void* {
    void* p = ws + off;
    off = (off + bytes + 255) & ~(size_t)255;
    return p;
  };
  unsigned short* W_e_cat   = (unsigned short*)alloc((size_t)kH4 * kXH * 2);
  unsigned short* W_d_cat   = (unsigned short*)alloc((size_t)kH4 * kXH * 2);
  unsigned short* W_ref_b   = (unsigned short*)alloc((size_t)kH * kH * 2);
  unsigned short* W_q_b     = (unsigned short*)alloc((size_t)kH * kH * 2);
  unsigned short* emb_b     = (unsigned short*)alloc((size_t)kB * kG * kE * 2);
  unsigned short* enc_out_b = (unsigned short*)alloc((size_t)kB * kG * kH * 2);
  unsigned short* ref_projb = (unsigned short*)alloc((size_t)kB * kG * kH * 2);
  unsigned short* x_b       = (unsigned short*)alloc((size_t)kB * kE * 2);
  unsigned short* h_b       = (unsigned short*)alloc((size_t)kB * kH * 2);
  float*          gates     = (float*)alloc((size_t)kB * kH4 * 4);
  float*          c_st      = (float*)alloc((size_t)kB * kH * 4);
  float*          q_st      = (float*)alloc((size_t)kB * kH * 4);
  float*          mask      = (float*)alloc((size_t)kB * kG * 4);

  const int TPB = 256;
  auto nb = [](int n) { return (n + 255) / 256; };

  // one-time packing / init
  pack_wcat<<<nb(kH4 * kXH), TPB, 0, stream>>>(Wih_e, Whh_e, W_e_cat);
  pack_wcat<<<nb(kH4 * kXH), TPB, 0, stream>>>(Wih_d, Whh_d, W_d_cat);
  f32_to_bf16_k<<<nb(kH * kH), TPB, 0, stream>>>(W_ref, W_ref_b, kH * kH);
  f32_to_bf16_k<<<nb(kH * kH), TPB, 0, stream>>>(W_q, W_q_b, kH * kH);
  embed_kernel<<<nb(kB * kG * kE), TPB, 0, stream>>>(state, node_emb, emb_b);
  fill_u16<<<nb(kB * kH), TPB, 0, stream>>>(h_b, 0, kB * kH);   // h0 = 0
  fill_f32<<<nb(kB * kH), TPB, 0, stream>>>(c_st, 0.0f, kB * kH);

  const dim3 gGates(kB / 64, kH4 / 128);             // (8, 16)
  const dim3 gRef((kB * kG) / 64, kH / 128);         // (1024, 4)
  const dim3 gQ(kB / 64, kH / 128);                  // (8, 4)

  // ---- encoder: 128 dependent LSTM steps; x read in place from emb slices
  for (int t = 0; t < kG; ++t) {
    gemm_bt_wmma<false><<<gGates, 256, 0, stream>>>(
        emb_b + (size_t)t * kE, kG * kE, kE,        // A1 = emb[:, t, :]
        h_b, kH, kH,                                // A2 = h(t-1)
        W_e_cat, kXH, gates, kH4);
    lstm_pointwise<<<nb(kB * kH), TPB, 0, stream>>>(
        gates, bih_e, bhh_e, c_st, h_b,
        enc_out_b + (size_t)t * kH, (size_t)kG * kH);
  }

  // ---- ref_proj = enc_out @ W_ref^T  (bf16 output; stays L2-resident)
  gemm_bt_wmma<true><<<gRef, 256, 0, stream>>>(
      enc_out_b, kH, kH, nullptr, 0, 0, W_ref_b, kH, ref_projb, kH);

  // ---- decoder init: x = 0, mask = 1; (h,c) carry over from encoder
  fill_u16<<<nb(kB * kE), TPB, 0, stream>>>(x_b, 0, kB * kE);
  fill_f32<<<nb(kB * kG), TPB, 0, stream>>>(mask, 1.0f, kB * kG);

  float* out_actions = (float*)d_out;                    // (B,G,1) as float
  float* out_attn    = (float*)d_out + (size_t)kB * kG;  // (B,G,G)

  // ---- decoder: 128 steps of GEMM -> pointwise -> q GEMM -> attention
  for (int t = 0; t < kG; ++t) {
    gemm_bt_wmma<false><<<gGates, 256, 0, stream>>>(
        x_b, kE, kE, h_b, kH, kH, W_d_cat, kXH, gates, kH4);
    lstm_pointwise<<<nb(kB * kH), TPB, 0, stream>>>(
        gates, bih_d, bhh_d, c_st, h_b, nullptr, 0);
    gemm_bt_wmma<false><<<gQ, 256, 0, stream>>>(
        h_b, kH, kH, nullptr, 0, 0, W_q_b, kH, q_st, kH);
    attention_step<<<kB, kG, 0, stream>>>(ref_projb, q_st, vvec, mask,
                                          action_emb, x_b,
                                          out_actions, out_attn, t);
  }
}
